// NonLocalAttention_5325759447373
// MI455X (gfx1250) — matrix operand order repaired and checked by reading, not compiled, not run on hardware
//
#include <hip/hip_runtime.h>
#include <hip/hip_bf16.h>

#define HW_    9216
#define NB     2
#define CIN    128
#define CE     64
#define KTILE  64
#define KB_    (HW_ / KTILE)   /* 144 key steps */
#define LOG2E  1.4426950408889634f

typedef __attribute__((ext_vector_type(16))) _Float16 v16h;
typedef __attribute__((ext_vector_type(8)))  _Float16 v8h;
typedef __attribute__((ext_vector_type(2)))  __fp16   v2fp;
typedef __attribute__((ext_vector_type(8)))  float    v8f;

// Load a 16x32 f16 A-fragment row chunk: elem e <- p[hi*16 + e]
// (contiguous per lane, matching B).
static __device__ inline v16h loadA(const _Float16* p, int hi) {
    v8h a = *(const v8h*)(p + hi * 16);
    v8h b = *(const v8h*)(p + hi * 16 + 8);
    return __builtin_shufflevector(a, b, 0,1,2,3,4,5,6,7,8,9,10,11,12,13,14,15);
}

// Load a 32x16 f16 B-fragment: elem e <- p[hi*16 + e] (contiguous per lane).
static __device__ inline v16h loadB(const _Float16* p, int hi) {
    v8h a = *(const v8h*)(p + hi * 16);
    v8h b = *(const v8h*)(p + hi * 16 + 8);
    return __builtin_shufflevector(a, b, 0,1,2,3,4,5,6,7,8,9,10,11,12,13,14,15);
}

static __device__ inline uint32_t pkh(float a, float b) {
    union { v2fp h2; uint32_t u; } t;
    t.h2 = __builtin_amdgcn_cvt_pkrtz(a, b);
    return t.u;
}

// Convert 16 f32 (two runs of 8) into an f16 A-fragment register pair.
static __device__ inline v16h cvtA16(const float* p0, const float* p1) {
    union { v16h h; uint32_t u[8]; } r;
#pragma unroll
    for (int d = 0; d < 4; ++d) r.u[d]     = pkh(p0[2*d], p0[2*d + 1]);
#pragma unroll
    for (int d = 0; d < 4; ++d) r.u[d + 4] = pkh(p1[2*d], p1[2*d + 1]);
    return r.h;
}

// Build P~ B-fragment (32 keys x 16 queries) from two packed C-layout tiles.
static __device__ inline v16h buildB(const uint32_t* a, const uint32_t* b, int hi) {
    union { v16h h; uint32_t u[8]; } r;
#pragma unroll
    for (int j = 0; j < 4; ++j) {
        uint32_t send = hi ? a[j] : b[j];
        uint32_t y = (uint32_t)__shfl_xor((int)send, 16);
        r.u[j]     = hi ? y    : a[j];
        r.u[j + 4] = hi ? b[j] : y;
    }
    return r.h;
}

// ---------------------------------------------------------------------------
// Kernel 1: WMMA 1x1 conv + PReLU.  Y[oc][p] = W[oc][c] X[c][p], 256 oc.
// Each wave owns 2 oc-tiles (W fragments in registers); block stages a
// 16-pixel x 128-channel X tile (f32->f16, transposed) in LDS.
// Writes q (scaled by log2e), k as [pixel][64]; v as [ch][pixel].
// ---------------------------------------------------------------------------
__global__ __launch_bounds__(256, 1) void nla_qkv(
    const float* __restrict__ x,
    const float* __restrict__ w1, const float* __restrict__ b1, const float* __restrict__ a1,
    const float* __restrict__ w2, const float* __restrict__ b2, const float* __restrict__ a2,
    const float* __restrict__ wa, const float* __restrict__ ba, const float* __restrict__ aa,
    _Float16* __restrict__ qb, _Float16* __restrict__ kb, _Float16* __restrict__ vb)
{
    __shared__ _Float16 sX[2][16][136];   // pixel-major, pad 128->136

    const int tid  = threadIdx.x;
    const int lane = tid & 31;
    const int wv   = tid >> 5;
    const int hi   = lane >> 4;
    const int col  = lane & 15;

    const int p0base = blockIdx.x * 128;          // 128 pixels per block
    const int n      = p0base / HW_;
    const int pl0    = p0base - n * HW_;
    const float* xn  = x + (size_t)n * CIN * HW_;

    // --- per-wave weight fragments: oc-tiles (wv*32) and (wv*32+16) ---
    v16h  aW[2][4];
    float bias_r[2][8], slope[2];
#pragma unroll
    for (int s = 0; s < 2; ++s) {
        const int oc0 = wv * 32 + s * 16;
        const float *wsrc, *bsrc, *asrc; int rel;
        if (oc0 < CE)        { wsrc = w1; bsrc = b1; asrc = a1; rel = oc0; }
        else if (oc0 < 2*CE) { wsrc = w2; bsrc = b2; asrc = a2; rel = oc0 - CE; }
        else                 { wsrc = wa; bsrc = ba; asrc = aa; rel = oc0 - 2*CE; }
        slope[s] = asrc[0];
        const float* wrow = wsrc + (size_t)(rel + col) * CIN;
#pragma unroll
        for (int k = 0; k < 4; ++k)
            aW[s][k] = cvtA16(wrow + 32*k + hi*16, wrow + 32*k + hi*16 + 8);
#pragma unroll
        for (int r = 0; r < 8; ++r) bias_r[s][r] = bsrc[rel + r + 8*hi];
    }

    auto stage = [&](int it, int buf) {
        const int c  = tid >> 1;
        const int ph = (tid & 1) * 8;
        const float* src = xn + (size_t)c * HW_ + pl0 + it * 16 + ph;
        float f[8];
#pragma unroll
        for (int j = 0; j < 8; ++j) f[j] = src[j];
#pragma unroll
        for (int j = 0; j < 8; ++j) sX[buf][ph + j][c] = (_Float16)f[j];
    };

    stage(0, 0);
    for (int it = 0; it < 8; ++it) {
        const int buf = it & 1;
        __syncthreads();
        if (it + 1 < 8) stage(it + 1, (it + 1) & 1);

        v16h bX[4];
#pragma unroll
        for (int k = 0; k < 4; ++k) bX[k] = loadB(&sX[buf][col][32 * k], hi);

#pragma unroll
        for (int s = 0; s < 2; ++s) {
            v8f acc;
#pragma unroll
            for (int r = 0; r < 8; ++r) acc[r] = 0.0f;
#pragma unroll
            for (int k = 0; k < 4; ++k)
                acc = __builtin_amdgcn_wmma_f32_16x16x32_f16(false, aW[s][k], false, bX[k], (short)0, acc, false, false);

            const int oc0 = wv * 32 + s * 16;
            const int p   = p0base + it * 16 + col;  // global pixel
            const int pl  = pl0 + it * 16 + col;     // pixel within batch
            float y[8];
#pragma unroll
            for (int r = 0; r < 8; ++r) {
                float v = acc[r] + bias_r[s][r];
                y[r] = (v >= 0.0f) ? v : slope[s] * v;
            }
            if (oc0 < CE) {                 // q: fold log2(e) for exp2 softmax
                v8h h;
#pragma unroll
                for (int r = 0; r < 8; ++r) h[r] = (_Float16)(y[r] * LOG2E);
                *(v8h*)&qb[(size_t)p * CE + oc0 + 8*hi] = h;
            } else if (oc0 < 2*CE) {        // k
                v8h h;
#pragma unroll
                for (int r = 0; r < 8; ++r) h[r] = (_Float16)y[r];
                *(v8h*)&kb[(size_t)p * CE + (oc0 - CE) + 8*hi] = h;
            } else {                        // v: channel-major
#pragma unroll
                for (int r = 0; r < 8; ++r)
                    vb[((size_t)n * CIN + (oc0 - 2*CE) + r + 8*hi) * HW_ + pl] = (_Float16)y[r];
            }
        }
    }
}

// ---------------------------------------------------------------------------
// Kernel 2: flash attention, transposed-S formulation.
//   S^T = K * Q^T   (keys in VGPR dim, queries in lanes)
//   O^T = V^T * P~  (channels in VGPR dim, queries in lanes)
// Scores arrive pre-scaled by log2(e) -> softmax uses exp2.
// ---------------------------------------------------------------------------
__global__ __launch_bounds__(256, 1) void nla_attn(
    const _Float16* __restrict__ qb,
    const _Float16* __restrict__ kbuf,
    const _Float16* __restrict__ vbuf,
    float* __restrict__ out)
{
    __shared__ _Float16 sK[2][KTILE][72];  // 64 keys x 64, pad->72 (144B stride)
    __shared__ _Float16 sV[2][CIN][72];    // V^T: 128 ch x 64 keys, pad->72

    const int tid  = threadIdx.x;
    const int lane = tid & 31;
    const int wv   = tid >> 5;
    const int hi   = lane >> 4;
    const int col  = lane & 15;

    const int    qt    = blockIdx.x * 8 + wv;
    const int    grow0 = qt * 16;
    const int    n     = grow0 / HW_;
    const int    pl0   = grow0 - n * HW_;
    const size_t nHW   = (size_t)n * HW_;
    const size_t nC    = (size_t)n * CIN;

    const _Float16* qr = qb + (size_t)(grow0 + col) * CE;
    v16h bQ0 = loadB(qr, hi);
    v16h bQ1 = loadB(qr + 32, hi);

    v8f O[8];
#pragma unroll
    for (int t = 0; t < 8; ++t)
#pragma unroll
        for (int r = 0; r < 8; ++r) O[t][r] = 0.0f;
    float mrun = -3.0e38f, lrun = 0.0f;

    auto stage = [&](int kb, int buf) {
        const int kr0 = kb * KTILE;
        {   // K tile: 64 x 64 f16, 16 per thread
            const int r  = tid >> 2;
            const int cc = (tid & 3) * 16;
            const _Float16* src = kbuf + (nHW + (size_t)(kr0 + r)) * CE + cc;
            v8h d0 = *(const v8h*)src;
            v8h d1 = *(const v8h*)(src + 8);
            *(v8h*)&sK[buf][r][cc]     = d0;
            *(v8h*)&sK[buf][r][cc + 8] = d1;
        }
        {   // V^T tile: 128 x 64 f16, 32 per thread
            const int r   = tid >> 1;
            const int off = (tid & 1) * 32;
            const _Float16* src = vbuf + (nC + r) * HW_ + kr0 + off;
            v8h d0 = *(const v8h*)src;
            v8h d1 = *(const v8h*)(src + 8);
            v8h d2 = *(const v8h*)(src + 16);
            v8h d3 = *(const v8h*)(src + 24);
            *(v8h*)&sV[buf][r][off]      = d0;
            *(v8h*)&sV[buf][r][off + 8]  = d1;
            *(v8h*)&sV[buf][r][off + 16] = d2;
            *(v8h*)&sV[buf][r][off + 24] = d3;
        }
    };

    stage(0, 0);

    for (int kb = 0; kb < KB_; ++kb) {
        const int buf = kb & 1;
        __syncthreads();
        if (kb + 1 < KB_) stage(kb + 1, (kb + 1) & 1);

        // ---- S^T = K @ Q^T : 4 tiles of (16 keys x 16 queries) ----
        v8f c[4];
#pragma unroll
        for (int kt = 0; kt < 4; ++kt) {
            const _Float16* kr = &sK[buf][kt * 16 + col][0];
            v16h aK0 = loadA(kr, hi);
            v16h aK1 = loadA(kr + 32, hi);
            v8f z;
#pragma unroll
            for (int r = 0; r < 8; ++r) z[r] = 0.0f;
            c[kt] = __builtin_amdgcn_wmma_f32_16x16x32_f16(false, aK0, false, bQ0, (short)0, z,     false, false);
            c[kt] = __builtin_amdgcn_wmma_f32_16x16x32_f16(false, aK1, false, bQ1, (short)0, c[kt], false, false);
        }

        // ---- online softmax over 64 keys (base-2 domain) ----
        float mx = c[0][0];
#pragma unroll
        for (int kt = 0; kt < 4; ++kt)
#pragma unroll
            for (int r = 0; r < 8; ++r) mx = fmaxf(mx, c[kt][r]);
        mx = fmaxf(mx, __shfl_xor(mx, 16));
        const float mn = fmaxf(mrun, mx);
        const float al = exp2f(mrun - mn);
        mrun = mn;
        float rs = 0.0f;
#pragma unroll
        for (int kt = 0; kt < 4; ++kt)
#pragma unroll
            for (int r = 0; r < 8; ++r) {
                c[kt][r] = exp2f(c[kt][r] - mn);
                rs += c[kt][r];
            }
        rs += __shfl_xor(rs, 16);
        lrun = lrun * al + rs;
#pragma unroll
        for (int t = 0; t < 8; ++t)
#pragma unroll
            for (int r = 0; r < 8; ++r) O[t][r] *= al;

        // ---- pack P~ to f16, build B-fragments ----
        uint32_t pk[4][4];
#pragma unroll
        for (int kt = 0; kt < 4; ++kt)
#pragma unroll
            for (int j = 0; j < 4; ++j)
                pk[kt][j] = pkh(c[kt][2*j], c[kt][2*j + 1]);
        v16h bP0 = buildB(pk[0], pk[1], hi);
        v16h bP1 = buildB(pk[2], pk[3], hi);

        // ---- O^T += V^T @ P~ : 8 channel tiles, K-dim 64 ----
#pragma unroll
        for (int t = 0; t < 8; ++t) {
            const _Float16* vr = &sV[buf][t * 16 + col][0];
            v16h aV0 = loadA(vr, hi);
            v16h aV1 = loadA(vr + 32, hi);
            O[t] = __builtin_amdgcn_wmma_f32_16x16x32_f16(false, aV0, false, bP0, (short)0, O[t], false, false);
            O[t] = __builtin_amdgcn_wmma_f32_16x16x32_f16(false, aV1, false, bP1, (short)0, O[t], false, false);
        }
    }

    const float inv = 1.0f / lrun;
#pragma unroll
    for (int t = 0; t < 8; ++t)
#pragma unroll
        for (int r = 0; r < 8; ++r) {
            const int ch = t * 16 + r + 8 * hi;
            out[(nC + ch) * HW_ + pl0 + col] = O[t][r] * inv;
        }
}

// ---------------------------------------------------------------------------
extern "C" void kernel_launch(void* const* d_in, const int* in_sizes, int n_in,
                              void* d_out, int out_size, void* d_ws, size_t ws_size,
                              hipStream_t stream) {
    const float* x  = (const float*)d_in[0];
    const float* w1 = (const float*)d_in[1];
    const float* b1 = (const float*)d_in[2];
    const float* a1 = (const float*)d_in[3];
    const float* w2 = (const float*)d_in[4];
    const float* b2 = (const float*)d_in[5];
    const float* a2 = (const float*)d_in[6];
    const float* wa = (const float*)d_in[7];
    const float* ba = (const float*)d_in[8];
    const float* aa = (const float*)d_in[9];

    _Float16* qb = (_Float16*)d_ws;
    _Float16* kb = qb + (size_t)NB * HW_ * CE;
    _Float16* vb = kb + (size_t)NB * HW_ * CE;

    nla_qkv<<<dim3((NB * HW_) / 128), 256, 0, stream>>>(
        x, w1, b1, a1, w2, b2, a2, wa, ba, aa, qb, kb, vb);

    nla_attn<<<dim3((NB * HW_) / (16 * 8)), 256, 0, stream>>>(
        qb, kb, vb, (float*)d_out);
}
